// AttentionBlock_56470230007905
// MI455X (gfx1250) — matrix operand order repaired and checked by reading, not compiled, not run on hardware
//
#include <hip/hip_runtime.h>

// MI455X / gfx1250, wave32. WMMA f16 path: v_wmma_f32_16x16x32_f16.
// d_ws layout: Q | K | V, each [B*T, 64] _Float16  -> 3 * 8192*64*2 = 3 MB.

typedef __attribute__((ext_vector_type(16))) _Float16 v16h;
typedef __attribute__((ext_vector_type(8)))  _Float16 v8h;
typedef __attribute__((ext_vector_type(8)))  float    v8f;
typedef __attribute__((ext_vector_type(4)))  int      v4i;

typedef __attribute__((address_space(1))) v4i glb_v4i;   // global 16B vector
typedef __attribute__((address_space(3))) v4i lds_v4i;   // LDS 16B vector

#define D_IN 1024
#define DK   64
#define T    2048
#define NB   4
#define BT   (NB * T)      // 8192 rows

#if __has_builtin(__builtin_amdgcn_global_load_async_to_lds_b128)
#define HAVE_ASYNC_LDS 1
#endif

static __device__ __forceinline__ v16h cat8(v8h lo, v8h hi) {
  return __builtin_shufflevector(lo, hi, 0,1,2,3,4,5,6,7,8,9,10,11,12,13,14,15);
}

// ---------------------------------------------------------------------------
// Kernel 1: fused QKV projection.
// Grid: 12 column tiles (Q:0-3,K:4-7,V:8-11) x 64 row groups = 768 blocks.
// All 8 waves of a block share one 16-wide W column panel, staged once in LDS
// pre-swizzled into B-fragment order; per K-step the B frag is 2 ds_load_b128.
// ---------------------------------------------------------------------------
__global__ __launch_bounds__(256) void qkv_proj_kernel(
    const float* __restrict__ x,  const float* __restrict__ Wq,
    const float* __restrict__ Wk, const float* __restrict__ Wv,
    _Float16* __restrict__ qkv) {
  // Wlds[(kstep, half, n, j)] = W[kstep*32 + half*16 + j][col0 + n]
  __shared__ _Float16 Wlds[32 * 512];        // 32 KB

  const int lane  = threadIdx.x & 31;
  const int wslot = threadIdx.x >> 5;
  const int ct      = blockIdx.x % 12;       // which 16-col panel of [Wq|Wk|Wv]
  const int rowTile = (blockIdx.x / 12) * 8 + wslot;
  const int mat     = ct >> 2;               // 0=Q 1=K 2=V
  const int col0    = (ct & 3) * 16;
  const float* W    = (mat == 0) ? Wq : (mat == 1) ? Wk : Wv;
  _Float16* Out     = qkv + (size_t)mat * BT * DK;

  // ---- cooperative W panel staging: 1024 rows x 16 cols, 4 rows per thread
  {
    const int k0 = threadIdx.x * 4;
    #pragma unroll
    for (int kk = 0; kk < 4; ++kk) {
      const int k = k0 + kk;
      const int kstep = k >> 5, half = (k >> 4) & 1, j = k & 15;
      const float4* wr = (const float4*)(W + (size_t)k * DK + col0);
      float4 w0 = wr[0], w1 = wr[1], w2 = wr[2], w3 = wr[3];
      _Float16* dst = &Wlds[kstep * 512 + half * 256 + j];
      dst[0*16]=(_Float16)w0.x; dst[1*16]=(_Float16)w0.y; dst[2*16]=(_Float16)w0.z; dst[3*16]=(_Float16)w0.w;
      dst[4*16]=(_Float16)w1.x; dst[5*16]=(_Float16)w1.y; dst[6*16]=(_Float16)w1.z; dst[7*16]=(_Float16)w1.w;
      dst[8*16]=(_Float16)w2.x; dst[9*16]=(_Float16)w2.y; dst[10*16]=(_Float16)w2.z; dst[11*16]=(_Float16)w2.w;
      dst[12*16]=(_Float16)w3.x; dst[13*16]=(_Float16)w3.y; dst[14*16]=(_Float16)w3.z; dst[15*16]=(_Float16)w3.w;
    }
  }
  __syncthreads();

  const int half_sel = lane >> 4;            // 0: lanes 0-15, 1: lanes 16-31
  const int nlane    = lane & 15;
  const int arow     = rowTile * 16 + nlane; // A fragment row = M

  v8f c = {};
  for (int kstep = 0; kstep < 32; ++kstep) {
    const int kb = kstep * 32;
    // ---- A fragment (x, f32->f16), 16x32 f16 layout:
    // lanes 0-15: K = {kb+0..7, kb+16..23}; lanes 16-31: K = {kb+8..15, kb+24..31}
    const int ka = kb + half_sel * 8;
    const float4* pa = (const float4*)(x + (size_t)arow * D_IN + ka);
    float4 f0 = pa[0], f1 = pa[1];
    const float4* pb = (const float4*)(x + (size_t)arow * D_IN + ka + 16);
    float4 f2 = pb[0], f3 = pb[1];
    v16h a;
    a[0]=(_Float16)f0.x;  a[1]=(_Float16)f0.y;  a[2]=(_Float16)f0.z;  a[3]=(_Float16)f0.w;
    a[4]=(_Float16)f1.x;  a[5]=(_Float16)f1.y;  a[6]=(_Float16)f1.z;  a[7]=(_Float16)f1.w;
    a[8]=(_Float16)f2.x;  a[9]=(_Float16)f2.y;  a[10]=(_Float16)f2.z; a[11]=(_Float16)f2.w;
    a[12]=(_Float16)f3.x; a[13]=(_Float16)f3.y; a[14]=(_Float16)f3.z; a[15]=(_Float16)f3.w;

    // ---- B fragment from pre-swizzled LDS panel: 2 x ds_load_b128
    const _Float16* bp = &Wlds[kstep * 512 + nlane * 16 + half_sel * 8];
    v16h bfrag = cat8(*(const v8h*)bp, *(const v8h*)(bp + 256));

    c = __builtin_amdgcn_wmma_f32_16x16x32_f16(false, a, false, bfrag,
                                               (short)0, c, false, false);
  }

  // C/D layout: VGPR r -> (M = r + 8*half_sel, N = nlane)
  #pragma unroll
  for (int r = 0; r < 8; ++r) {
    const int M = r + half_sel * 8;
    Out[(size_t)(rowTile * 16 + M) * DK + col0 + nlane] = (_Float16)c[r];
  }
}

// ---------------------------------------------------------------------------
// Kernel 2: causal flash attention, one wave per 16-query tile.
// 4 batches * 128 qtiles = 512 waves = 64 blocks of 256 threads (8 waves).
// ---------------------------------------------------------------------------
__global__ __launch_bounds__(256) void flash_attn_kernel(
    const _Float16* __restrict__ qkv, float* __restrict__ out) {
  __shared__ _Float16 Pst[8][16 * 32];   // per-wave P transpose staging (1 KB)
  __shared__ _Float16 Vst[8][32 * DK];   // per-wave V chunk staging  (4 KB)

  const int lane  = threadIdx.x & 31;
  const int wslot = threadIdx.x >> 5;
  const int wid   = blockIdx.x * 8 + wslot;

  const int batch = wid >> 7;            // 128 qtiles per batch
  const int q0    = (wid & 127) * 16;

  const _Float16* Qb = qkv + (size_t)batch * T * DK;
  const _Float16* Kb = qkv + (size_t)BT * DK + (size_t)batch * T * DK;
  const _Float16* Vb = qkv + (size_t)2 * BT * DK + (size_t)batch * T * DK;

  const int half_sel = lane >> 4;
  const int nlane    = lane & 15;
  const int d0       = half_sel * 8;     // A-fragment K-run start (0 or 8)
  const int qrow     = q0 + nlane;

  // Q as two A fragments (d_k = 64 = 2 x K32), loaded once.
  v16h aq0 = cat8(*(const v8h*)(Qb + (size_t)qrow * DK + d0),
                  *(const v8h*)(Qb + (size_t)qrow * DK + d0 + 16));
  v16h aq1 = cat8(*(const v8h*)(Qb + (size_t)qrow * DK + 32 + d0),
                  *(const v8h*)(Qb + (size_t)qrow * DK + 32 + d0 + 16));

  float m[8], l[8];
  #pragma unroll
  for (int r = 0; r < 8; ++r) { m[r] = -__builtin_inff(); l[r] = 0.0f; }
  v8f acc[4] = {{}, {}, {}, {}};

  const int nk = q0 + 16;                // keys needed: 0 .. q0+15
  for (int kt0 = 0; kt0 < nk; kt0 += 32) {
    // ---- stage V chunk [kt0 .. kt0+31] x 64 into LDS (coalesced rows)
    {
      const _Float16* vr = Vb + (size_t)(kt0 + lane) * DK;
      _Float16* vd = &Vst[wslot][lane * DK];
#ifdef HAVE_ASYNC_LDS
      #pragma unroll
      for (int j = 0; j < 8; ++j)
        __builtin_amdgcn_global_load_async_to_lds_b128(
            (glb_v4i*)(vr + j * 8), (lds_v4i*)(vd + j * 8), 0, 0);
#else
      #pragma unroll
      for (int j = 0; j < 8; ++j)
        *(v8h*)(vd + j * 8) = *(const v8h*)(vr + j * 8);
#endif
    }
    // prefetch next chunk of K/V while this one computes
    if (kt0 + 32 < nk) {
      __builtin_prefetch(Kb + (size_t)(kt0 + 32 + lane) * DK, 0, 3);
      __builtin_prefetch(Vb + (size_t)(kt0 + 32 + lane) * DK, 0, 3);
    }

    // ---- scores: two 16x16 S tiles (keys t*16 .. t*16+15), 2 WMMAs each
    v8f s[2];
    #pragma unroll
    for (int t = 0; t < 2; ++t) {
      const _Float16* kr = Kb + (size_t)(kt0 + t * 16 + nlane) * DK;
      // B fragment of K^T: lanes 0-15 hold d={0..7,16..23}, lanes 16-31 d={8..15,24..31}
      v16h b0 = cat8(*(const v8h*)(kr + half_sel * 8),
                     *(const v8h*)(kr + half_sel * 8 + 16));
      v16h b1 = cat8(*(const v8h*)(kr + 32 + half_sel * 8),
                     *(const v8h*)(kr + 32 + half_sel * 8 + 16));
      v8f z = {};
      v8f st = __builtin_amdgcn_wmma_f32_16x16x32_f16(false, aq0, false, b0,
                                                      (short)0, z, false, false);
      st = __builtin_amdgcn_wmma_f32_16x16x32_f16(false, aq1, false, b1,
                                                  (short)0, st, false, false);
      s[t] = st;
    }

    // ---- causal mask (diagonal chunks only)
    if (kt0 + 31 > q0) {
      #pragma unroll
      for (int r = 0; r < 8; ++r) {
        const int q = q0 + r + half_sel * 8;
        if (kt0 + nlane > q)      s[0][r] = -__builtin_inff();
        if (kt0 + 16 + nlane > q) s[1][r] = -__builtin_inff();
      }
    }

    // ---- online softmax (per-row stats via half-wave butterflies)
    float p0[8], p1[8], sc[8];
    #pragma unroll
    for (int r = 0; r < 8; ++r) {
      float mx = fmaxf(s[0][r], s[1][r]);
      mx = fmaxf(mx, __shfl_xor(mx, 8, 16));
      mx = fmaxf(mx, __shfl_xor(mx, 4, 16));
      mx = fmaxf(mx, __shfl_xor(mx, 2, 16));
      mx = fmaxf(mx, __shfl_xor(mx, 1, 16));
      const float mnew = fmaxf(m[r], mx);
      sc[r] = __expf(m[r] - mnew);
      p0[r] = __expf(s[0][r] - mnew);
      p1[r] = __expf(s[1][r] - mnew);
      float rs = p0[r] + p1[r];
      rs += __shfl_xor(rs, 8, 16);
      rs += __shfl_xor(rs, 4, 16);
      rs += __shfl_xor(rs, 2, 16);
      rs += __shfl_xor(rs, 1, 16);
      l[r] = l[r] * sc[r] + rs;
      m[r] = mnew;
    }
    #pragma unroll
    for (int nt = 0; nt < 4; ++nt)
      #pragma unroll
      for (int r = 0; r < 8; ++r)
        acc[nt][r] *= sc[r];

    // ---- P (C-layout) -> LDS -> A-fragment (16x32 f16)
    #pragma unroll
    for (int r = 0; r < 8; ++r) {
      const int M = r + half_sel * 8;
      Pst[wslot][M * 32 + nlane]      = (_Float16)p0[r];
      Pst[wslot][M * 32 + 16 + nlane] = (_Float16)p1[r];
    }
#ifdef HAVE_ASYNC_LDS
    asm volatile("s_wait_asynccnt 0" ::: "memory");  // V chunk landed in LDS
#endif
    asm volatile("s_wait_dscnt 0" ::: "memory");     // wave-local LDS fence
    v16h ap = cat8(*(const v8h*)&Pst[wslot][nlane * 32 + d0],
                   *(const v8h*)&Pst[wslot][nlane * 32 + d0 + 16]);

    // ---- AV: 4 output column tiles, B fragment from staged V
    const int kv0 = half_sel * 8;      // lanes 0-15: k={0..7,16..23}, lanes 16-31: {8..15,24..31}
    #pragma unroll
    for (int nt = 0; nt < 4; ++nt) {
      v16h bv;
      #pragma unroll
      for (int j = 0; j < 16; ++j)
        bv[j] = Vst[wslot][(kv0 + (j & 7) + (j >> 3) * 16) * DK + nt * 16 + nlane];
      acc[nt] = __builtin_amdgcn_wmma_f32_16x16x32_f16(false, ap, false, bv,
                                                       (short)0, acc[nt], false, false);
    }
  }

  // ---- finalize: out = acc / (l * sqrt(64))   (post-softmax divide, per ref)
  float* ob = out + (size_t)batch * T * DK;
  #pragma unroll
  for (int r = 0; r < 8; ++r) {
    const int M = r + half_sel * 8;
    const float inv = 1.0f / (l[r] * 8.0f);
    #pragma unroll
    for (int nt = 0; nt < 4; ++nt)
      ob[(size_t)(q0 + M) * DK + nt * 16 + nlane] = acc[nt][r] * inv;
  }
}

// ---------------------------------------------------------------------------
extern "C" void kernel_launch(void* const* d_in, const int* in_sizes, int n_in,
                              void* d_out, int out_size, void* d_ws, size_t ws_size,
                              hipStream_t stream) {
  (void)in_sizes; (void)n_in; (void)out_size; (void)ws_size;
  const float* x  = (const float*)d_in[0];
  const float* Wq = (const float*)d_in[1];
  const float* Wk = (const float*)d_in[2];
  const float* Wv = (const float*)d_in[3];
  _Float16* qkv = (_Float16*)d_ws;        // 3 MB scratch: Q|K|V f16
  float* out = (float*)d_out;

  // 12 column panels * 64 row groups = 768 blocks of 8 waves
  qkv_proj_kernel<<<768, 256, 0, stream>>>(x, Wq, Wk, Wv, qkv);
  // 4 * 128 = 512 waves / 8 = 64 blocks
  flash_attn_kernel<<<64, 256, 0, stream>>>(qkv, out);
}
